// AudioSae_86990267613471
// MI455X (gfx1250) — matrix operand order repaired and checked by reading, not compiled, not run on hardware
//
#include <hip/hip_runtime.h>

typedef __attribute__((ext_vector_type(16))) __bf16 bf16x16;
typedef __attribute__((ext_vector_type(8)))  __bf16 bf16x8;
typedef __attribute__((ext_vector_type(8)))  float  f32x8;

#define NB      8192
#define DIN     768
#define DHID    12288
#define KSEL    64

__device__ __forceinline__ unsigned short bfbits(float f) {
    unsigned u = __float_as_uint(f);
    unsigned r = u + 0x7FFFu + ((u >> 16) & 1u);   // RNE
    return (unsigned short)(r >> 16);
}

__device__ __forceinline__ unsigned flipKey(float f) {
    unsigned u = __float_as_uint(f);
    return (u & 0x80000000u) ? ~u : (u | 0x80000000u);  // monotone: bigger float -> bigger key
}

// ---------------- prep: xc = bf16(x - pre_bias) ----------------
__global__ __launch_bounds__(256)
void prep_x_kernel(const float* __restrict__ x, const float* __restrict__ pre_bias,
                   unsigned short* __restrict__ xc) {
    size_t base = ((size_t)blockIdx.x * 256 + threadIdx.x) * 4;
    if (base >= (size_t)NB * DIN) return;
    float4 v  = *(const float4*)(x + base);
    int   k   = (int)(base % DIN);
    float4 pb = *(const float4*)(pre_bias + k);
    ushort4 o;
    o.x = bfbits(v.x - pb.x); o.y = bfbits(v.y - pb.y);
    o.z = bfbits(v.z - pb.z); o.w = bfbits(v.w - pb.w);
    *(ushort4*)(xc + base) = o;
}

// ---------------- prep: we = bf16(W_enc) ----------------
__global__ __launch_bounds__(256)
void prep_w_kernel(const float* __restrict__ w, unsigned short* __restrict__ wb) {
    size_t base = ((size_t)blockIdx.x * 256 + threadIdx.x) * 4;
    if (base >= (size_t)DHID * DIN) return;
    float4 v = *(const float4*)(w + base);
    ushort4 o;
    o.x = bfbits(v.x); o.y = bfbits(v.y); o.z = bfbits(v.z); o.w = bfbits(v.w);
    *(ushort4*)(wb + base) = o;
}

// ---------------- prep: WdT[j][i] = W_dec[i][j] ----------------
__global__ void transpose_wdec_kernel(const float* __restrict__ wdec, float* __restrict__ wdt) {
    __shared__ float tile[32][33];
    int j0 = blockIdx.x * 32;   // DHID
    int i0 = blockIdx.y * 32;   // DIN
    for (int s = threadIdx.y; s < 32; s += 8)
        tile[s][threadIdx.x] = wdec[(size_t)(i0 + s) * DHID + (j0 + threadIdx.x)];
    __syncthreads();
    for (int s = threadIdx.y; s < 32; s += 8)
        wdt[(size_t)(j0 + s) * DIN + (i0 + threadIdx.x)] = tile[threadIdx.x][s];
}

// ---------------- encode GEMM: pre_acts = xc @ we^T + latent_bias (bf16 WMMA) ----------------
// block tile 128(M) x 256(N) x 32(K); 8 waves (2x4); wave tile 64x64 as 4x4 of 16x16
// LDS double-buffered; one barrier per K-step; global loads register-staged.
__global__ __launch_bounds__(256)
void encode_gemm_kernel(const unsigned short* __restrict__ xc,
                        const unsigned short* __restrict__ we,
                        const float* __restrict__ latent_bias,
                        float* __restrict__ pre_acts) {
    __shared__ __align__(16) unsigned short sA[2][128 * 40];
    __shared__ __align__(16) unsigned short sB[2][256 * 40];

    const int tid    = threadIdx.x;
    const int lane   = tid & 31;
    const int wid    = tid >> 5;
    const int waveM  = wid >> 2;          // 0..1 -> 64 rows each
    const int waveN  = wid & 3;           // 0..3 -> 64 cols each
    const int mBase  = blockIdx.y * 128;
    const int nBase  = blockIdx.x * 256;
    const int halfKA = (lane >> 4) * 8;   // A: chunks at halfKA and halfKA+16 (elements)
    const int halfKB = (lane >> 4) * 8;   // B: chunks at halfKB and halfKB+16 (elements)
    const int l16    = lane & 15;

    f32x8 acc[4][4];
    for (int ns = 0; ns < 4; ++ns) {
        float lb = latent_bias[nBase + waveN * 64 + ns * 16 + l16];
        f32x8 c;
        for (int v = 0; v < 8; ++v) c[v] = lb;
        for (int ms = 0; ms < 4; ++ms) acc[ms][ns] = c;
    }

    const int ldRow = tid >> 2;          // 0..63 (+64/128/192 strides)
    const int ldSeg = (tid & 3) * 8;     // element offset 0,8,16,24
    const unsigned short* aPtr = xc + (size_t)(mBase + ldRow) * DIN + ldSeg;
    const unsigned short* bPtr = we + (size_t)(nBase + ldRow) * DIN + ldSeg;

    uint4 ar[2], br[4];
    // prologue: K-step 0 into buffer 0
    for (int i = 0; i < 2; ++i) ar[i] = *(const uint4*)(aPtr + (size_t)i * 64 * DIN);
    for (int i = 0; i < 4; ++i) br[i] = *(const uint4*)(bPtr + (size_t)i * 64 * DIN);
    for (int i = 0; i < 2; ++i) *(uint4*)(sA[0] + (ldRow + i * 64) * 40 + ldSeg) = ar[i];
    for (int i = 0; i < 4; ++i) *(uint4*)(sB[0] + (ldRow + i * 64) * 40 + ldSeg) = br[i];
    __syncthreads();

    const int KSTEPS = DIN / 32;
    for (int kk = 0; kk < KSTEPS; ++kk) {
        const int  cur  = kk & 1;
        const bool more = (kk + 1) < KSTEPS;
        if (more) {
            const int k1 = (kk + 1) * 32;
            for (int i = 0; i < 2; ++i) ar[i] = *(const uint4*)(aPtr + (size_t)i * 64 * DIN + k1);
            for (int i = 0; i < 4; ++i) br[i] = *(const uint4*)(bPtr + (size_t)i * 64 * DIN + k1);
        }
        if (kk + 2 < KSTEPS) {   // speculative L2 prefetch of the K+2 tile
            __builtin_prefetch(aPtr + (kk + 2) * 32, 0, 3);
            __builtin_prefetch(bPtr + (kk + 2) * 32, 0, 3);
        }

        bf16x16 bfr[4];
        for (int ns = 0; ns < 4; ++ns) {
            const unsigned short* p = sB[cur] + (waveN * 64 + ns * 16 + l16) * 40 + halfKB;
            bf16x8 lo = *(const bf16x8*)(p);
            bf16x8 hi = *(const bf16x8*)(p + 16);
            bfr[ns] = __builtin_shufflevector(lo, hi, 0,1,2,3,4,5,6,7,8,9,10,11,12,13,14,15);
        }
        for (int ms = 0; ms < 4; ++ms) {
            const unsigned short* p = sA[cur] + (waveM * 64 + ms * 16 + l16) * 40 + halfKA;
            bf16x8 lo = *(const bf16x8*)(p);
            bf16x8 hi = *(const bf16x8*)(p + 16);
            bf16x16 af = __builtin_shufflevector(lo, hi, 0,1,2,3,4,5,6,7,8,9,10,11,12,13,14,15);
            for (int ns = 0; ns < 4; ++ns) {
                acc[ms][ns] = __builtin_amdgcn_wmma_f32_16x16x32_bf16(
                    false, af, false, bfr[ns], (short)0, acc[ms][ns], false, false);
            }
        }

        if (more) {
            const int nxt = cur ^ 1;
            for (int i = 0; i < 2; ++i) *(uint4*)(sA[nxt] + (ldRow + i * 64) * 40 + ldSeg) = ar[i];
            for (int i = 0; i < 4; ++i) *(uint4*)(sB[nxt] + (ldRow + i * 64) * 40 + ldSeg) = br[i];
            __syncthreads();
        }
    }

    for (int ms = 0; ms < 4; ++ms) {
        int mRow0 = mBase + waveM * 64 + ms * 16 + (lane >> 4) * 8;
        for (int ns = 0; ns < 4; ++ns) {
            int col = nBase + waveN * 64 + ns * 16 + l16;
            float* dst = pre_acts + (size_t)mRow0 * DHID + col;
            for (int v = 0; v < 8; ++v)
                dst[(size_t)v * DHID] = acc[ms][ns][v];
        }
    }
}

// ---------------- top-k + relu + rms-norm + decode, one block per row ----------------
__global__ __launch_bounds__(256)
void topk_norm_decode_kernel(const float* __restrict__ pre_acts,
                             const float* __restrict__ x,
                             const float* __restrict__ pre_bias,
                             const float* __restrict__ wdt,
                             float* __restrict__ xhat,
                             float* __restrict__ resid,
                             float* __restrict__ hOut,
                             float* __restrict__ fOut,
                             int* __restrict__ idxOut) {
    __shared__ __align__(16) float sRow[DHID];
    __shared__ unsigned sHist[256];
    __shared__ unsigned long long sCand[256];
    __shared__ int   sWIdx[KSEL];
    __shared__ float sWVal[KSEL];
    __shared__ float sWF[KSEL];
    __shared__ unsigned sCount, sBin1, sAbove1, sThr16;
    __shared__ float sSumSq, sScale;

    const int tid = threadIdx.x;
    const size_t row = blockIdx.x;
    const float* pr = pre_acts + row * (size_t)DHID;

    // stage row in LDS
    for (int it = 0; it < 12; ++it) {
        int j = (it * 256 + tid) * 4;
        *(float4*)(sRow + j) = *(const float4*)(pr + j);
    }
    sHist[tid] = 0;
    if (tid == 0) { sCount = 0; sSumSq = 0.f; }
    __syncthreads();

    // level-1 histogram on top 8 bits of order-preserving key
    for (int j = tid; j < DHID; j += 256)
        atomicAdd(&sHist[flipKey(sRow[j]) >> 24], 1u);
    __syncthreads();
    if (tid == 0) {
        unsigned cum = 0; int b = 255;
        for (; b > 0; --b) { cum += sHist[b]; if (cum >= (unsigned)KSEL) break; }
        if (cum < (unsigned)KSEL) cum += sHist[0];
        sBin1 = (unsigned)b;
        sAbove1 = cum - sHist[b];
    }
    __syncthreads();
    unsigned bin1 = sBin1, above1 = sAbove1;

    // level-2 histogram on next 8 bits within the boundary bin
    sHist[tid] = 0;
    __syncthreads();
    for (int j = tid; j < DHID; j += 256) {
        unsigned k = flipKey(sRow[j]);
        if ((k >> 24) == bin1) atomicAdd(&sHist[(k >> 16) & 0xFFu], 1u);
    }
    __syncthreads();
    if (tid == 0) {
        unsigned need = (unsigned)KSEL - above1;
        unsigned cum = 0; int b = 255;
        for (; b > 0; --b) { cum += sHist[b]; if (cum >= need) break; }
        if (cum < need) cum += sHist[0];
        sThr16 = (bin1 << 8) | (unsigned)b;
    }
    __syncthreads();
    unsigned thr16 = sThr16;

    // collect candidates (>= 16-bit threshold bucket); count >= 64 by construction
    for (int j = tid; j < DHID; j += 256) {
        unsigned k = flipKey(sRow[j]);
        if ((k >> 16) >= thr16) {
            unsigned p = atomicAdd(&sCount, 1u);
            if (p < 256u)
                sCand[p] = ((unsigned long long)k << 32) |
                           (unsigned long long)(0xFFFFFFFFu - (unsigned)j);  // tie: lower idx first
        }
    }
    __syncthreads();
    unsigned cnt = sCount; if (cnt > 256u) cnt = 256u;
    if ((unsigned)tid >= cnt) sCand[tid] = 0ull;
    __syncthreads();

    // bitonic sort, descending (256 keys, 256 threads)
    for (int ksz = 2; ksz <= 256; ksz <<= 1) {
        for (int j = ksz >> 1; j > 0; j >>= 1) {
            int ixj = tid ^ j;
            if (ixj > tid) {
                unsigned long long a = sCand[tid], b = sCand[ixj];
                bool dir = ((tid & ksz) == 0);
                if ((a < b) == dir) { sCand[tid] = b; sCand[ixj] = a; }
            }
            __syncthreads();
        }
    }

    // winners: relu, sumsq, indices
    if (tid < KSEL) {
        unsigned long long c = sCand[tid];
        int idx = (int)(0xFFFFFFFFu - (unsigned)(c & 0xFFFFFFFFull));
        sWIdx[tid] = idx;
        float v  = sRow[idx];
        float hv = v > 0.f ? v : 0.f;
        sWVal[tid] = hv;
        atomicAdd(&sSumSq, hv * hv);
        idxOut[row * KSEL + tid] = idx;
    }
    __syncthreads();
    if (tid == 0) sScale = rsqrtf(sSumSq / (float)DHID + 1e-8f);
    __syncthreads();
    float scale = sScale;

    // rebuild sRow as dense h
    for (int it = 0; it < 12; ++it) {
        int j = (it * 256 + tid) * 4;
        *(float4*)(sRow + j) = make_float4(0.f, 0.f, 0.f, 0.f);
    }
    __syncthreads();
    if (tid < KSEL) {
        sRow[sWIdx[tid]] = sWVal[tid];
        sWF[tid] = sWVal[tid] * scale;
    }
    __syncthreads();

    // stream h and f
    float* hRow = hOut + row * (size_t)DHID;
    float* fRow = fOut + row * (size_t)DHID;
    for (int it = 0; it < 12; ++it) {
        int j = (it * 256 + tid) * 4;
        float4 hv = *(const float4*)(sRow + j);
        *(float4*)(hRow + j) = hv;
        *(float4*)(fRow + j) = make_float4(hv.x * scale, hv.y * scale, hv.z * scale, hv.w * scale);
    }

    // sparse decode: x_hat = pre_bias + sum_t f_t * WdT[idx_t][:]; residual = x - x_hat
    float a0 = pre_bias[tid], a1 = pre_bias[tid + 256], a2 = pre_bias[tid + 512];
    for (int t = 0; t < KSEL; ++t) {
        float w = sWF[t];
        const float* wr = wdt + (size_t)sWIdx[t] * DIN;
        a0 += w * wr[tid];
        a1 += w * wr[tid + 256];
        a2 += w * wr[tid + 512];
    }
    const float* xr = x + row * (size_t)DIN;
    float* xh = xhat  + row * (size_t)DIN;
    float* rs = resid + row * (size_t)DIN;
    xh[tid]       = a0;  xh[tid + 256] = a1;  xh[tid + 512] = a2;
    rs[tid]       = xr[tid]       - a0;
    rs[tid + 256] = xr[tid + 256] - a1;
    rs[tid + 512] = xr[tid + 512] - a2;
}

extern "C" void kernel_launch(void* const* d_in, const int* in_sizes, int n_in,
                              void* d_out, int out_size, void* d_ws, size_t ws_size,
                              hipStream_t stream) {
    (void)in_sizes; (void)n_in; (void)out_size; (void)ws_size;
    const float* x           = (const float*)d_in[0];
    const float* W_enc       = (const float*)d_in[1];
    const float* W_dec       = (const float*)d_in[2];
    const float* pre_bias    = (const float*)d_in[3];
    const float* latent_bias = (const float*)d_in[4];
    float* out = (float*)d_out;

    // workspace layout
    unsigned short* xc = (unsigned short*)d_ws;                 // NB*DIN bf16
    unsigned short* we = xc + (size_t)NB * DIN;                 // DHID*DIN bf16
    float*          wdt = (float*)(we + (size_t)DHID * DIN);    // DHID*DIN f32

    // output layout (tuple concatenated): x_hat, residual, h, f, indices, pre_acts
    float* xhat   = out;
    float* resid  = out + (size_t)NB * DIN;
    float* hOut   = out + (size_t)2 * NB * DIN;
    float* fOut   = hOut + (size_t)NB * DHID;
    int*   idxOut = (int*)(fOut + (size_t)NB * DHID);
    float* preA   = (float*)(idxOut + (size_t)NB * KSEL);

    prep_x_kernel<<<(NB * DIN / 4 + 255) / 256, 256, 0, stream>>>(x, pre_bias, xc);
    prep_w_kernel<<<(DHID * DIN / 4 + 255) / 256, 256, 0, stream>>>(W_enc, we);
    transpose_wdec_kernel<<<dim3(DHID / 32, DIN / 32), dim3(32, 8), 0, stream>>>(W_dec, wdt);
    encode_gemm_kernel<<<dim3(DHID / 256, NB / 128), 256, 0, stream>>>(xc, we, latent_bias, preA);
    topk_norm_decode_kernel<<<NB, 256, 0, stream>>>(preA, x, pre_bias, wdt,
                                                    xhat, resid, hOut, fOut, idxOut);
}
